// ATTBase_60189671686453
// MI455X (gfx1250) — compile-verified, matrix operand order checked by reading, not run on hardware
//
#include <hip/hip_runtime.h>
#include <hip/hip_bf16.h>

#define DEVI __device__ __forceinline__

typedef __attribute__((ext_vector_type(16))) _Float16 v16h;
typedef __attribute__((ext_vector_type(8)))  float    v8f;
typedef __attribute__((ext_vector_type(2)))  _Float16 h2;

// ---------------- problem constants (from reference) ----------------
static constexpr int Hh    = 256;
static constexpr int Bb    = 4096;
static constexpr int ADV   = 4, GOOD = 4, BOXN = 9, RAMPN = 2;
static constexpr int AGENT = ADV + GOOD;          // 8
static constexpr int NOTH  = AGENT - 1;           // 7
static constexpr int OBS   = 10 + 10*(AGENT-1) + 13*BOXN + 12*RAMPN; // 221

// output layout: value[4096] | hidden_actor[4096*256] | rnn_hxs[4096]
static constexpr size_t OUT_HID = Bb;
static constexpr size_t OUT_RNN = Bb + (size_t)Bb*Hh;

// ---------------- workspace layout (offsets in _Float16 units) ------
static constexpr size_t SET_ADV  = 0;
static constexpr size_t SET_GOOD = 65536;
static constexpr size_t SET_BOX  = 131072;
static constexpr size_t SET_RAMP = 196608;
static constexpr size_t SET_FCW  = 262144;
static constexpr size_t SET_E1W  = 327680;   // 1280*256
static constexpr size_t SET_E2W  = 655360;
static constexpr size_t SET_SZ   = 720896;
static constexpr size_t ACT_BASE = 0;
static constexpr size_t ENC_BASE = SET_SZ;
static constexpr size_t CRT_CORR = 2*SET_SZ;
static constexpr size_t CRT_CFC  = 2*SET_SZ + 65536;
static constexpr size_t CRT_C1   = 2*SET_SZ + 131072; // 512*256
static constexpr size_t WS_FII_H = (size_t(4)<<20) / 2;  // byte 4MB
static constexpr size_t WS_FIJ_H = (size_t(6)<<20) / 2;  // byte 6MB

// LDS strides (pad to shift banks per row: +4 dwords per row)
static constexpr int LDA_H   = 264;   // f16 row stride for 256-wide bufs
static constexpr int LDA_QC  = 260;   // f32 row stride
static constexpr int LDA_CAT = 1288;  // f16 row stride for cat (1280 wide)

// ---------------- WMMA 16x16 tile helpers ---------------------------
// A: f16 [16][lda] row-major in LDS.  Bt: f16 [N][K] (n-major) in global.
// 16-bit fragment layout (ISA 7.12.2): lane&15 = M (A) / N (B),
// VGPR v holds pair at k = (v&3)*2 + (lane>>4)*8 + (v>>2)*16.
DEVI v8f wmma_acc(const _Float16* __restrict__ Arow,
                  const _Float16* __restrict__ Brow,
                  int K, int khalf, v8f acc)
{
  for (int kb = 0; kb < K; kb += 32) {
    v16h a, b;
#pragma unroll
    for (int v = 0; v < 8; ++v) {
      int k = kb + ((v & 3) << 1) + ((v >> 2) << 4) + khalf;
      h2 pa = *(const h2*)(Arow + k);
      h2 pb = *(const h2*)(Brow + k);
      a[2*v] = pa[0]; a[2*v+1] = pa[1];
      b[2*v] = pb[0]; b[2*v+1] = pb[1];
    }
    acc = __builtin_amdgcn_wmma_f32_16x16x32_f16(false, a, false, b,
                                                 (short)0, acc, false, false);
  }
  return acc;
}

// Full [16 x 256] (or [16 x Nout]) layer with 8 waves, 2 N-tiles per wave.
// outKind: 0 -> float* dst, 1 -> _Float16* dst.  ldd in elements.
DEVI void gemm_layer(const _Float16* __restrict__ A, int lda,
                     const _Float16* __restrict__ Bt, int ldb, int K,
                     const float* __restrict__ bias, bool do_tanh,
                     int outKind, void* dst, int ldd)
{
  const int tid   = threadIdx.x;
  const int wave  = tid >> 5;
  const int lane  = tid & 31;
  const int half  = lane >> 4;
  const int mr    = lane & 15;
  const int khalf = half << 3;
  const _Float16* Arow = A + (size_t)mr * lda;
#pragma unroll
  for (int i = 0; i < 2; ++i) {
    const int ncol = ((wave << 1) + i) << 4;
    const _Float16* Brow = Bt + (size_t)(ncol + mr) * ldb;
    v8f acc = {};
    acc = wmma_acc(Arow, Brow, K, khalf, acc);
    const int n     = ncol + mr;
    const float bi  = bias ? bias[n] : 0.f;
    const int mbase = half * 8;
#pragma unroll
    for (int r = 0; r < 8; ++r) {
      float v = acc[r] + bi;
      if (do_tanh) v = tanhf(v);
      if (outKind == 0) ((float*)dst)[(size_t)(mbase + r) * ldd + n] = v;
      else        ((_Float16*)dst)[(size_t)(mbase + r) * ldd + n] = (_Float16)v;
    }
  }
}

// ---------------- encoder parameter bundle --------------------------
struct EncP {
  const float* selfW; const float* selfB;
  const float* gW[4]; const float* gB[4];   // seeker, hider, box, ramp
  const float* fcB;   const float* enc1B;   const float* enc2B;
  const _Float16* corr[4];                  // f16, [N][K] transposed
  const _Float16* fcWt; const _Float16* enc1Wt; const _Float16* enc2Wt;
};

DEVI const float* row_ptr(int job, int rowbase, int m,
                          const float* __restrict__ inputs,
                          const float* __restrict__ share, int agent_i)
{
  int r = rowbase + m;
  if (job != 2) return inputs + (size_t)r * OBS;
  int bb = r / NOTH, nn = r - bb * NOTH;
  int ag = (nn < agent_i) ? nn : nn + 1;
  return share + ((size_t)bb * AGENT + ag) * OBS;
}

// ---------------- encoder kernel ------------------------------------
// blocks 0..255: actor(inputs) -> d_out hidden (f32)
// blocks 256..511: encoder(inputs) -> ws f_ii (f16)
// blocks 512..2303: encoder(others) -> ws f_ij (f16)
__global__ void __launch_bounds__(256)
enc_kernel(const float* __restrict__ inputs, const float* __restrict__ share,
           EncP actorP, EncP encP, const int* __restrict__ agent_i_p,
           float* __restrict__ outHidden, _Float16* __restrict__ fii,
           _Float16* __restrict__ fij)
{
  __shared__ _Float16 sh_self[16 * LDA_H];
  __shared__ _Float16 sh_ent [16 * LDA_H];
  __shared__ float    sh_qc  [16 * LDA_QC];
  __shared__ _Float16 sh_cat [16 * LDA_CAT];
  __shared__ _Float16 sh_h   [16 * LDA_H];
  __shared__ float    sh_red [256];
  __shared__ float    sh_beta [16 * 9];
  __shared__ float    sh_alpha[16 * 9];

  const int tile = blockIdx.x;
  int job, rowbase;
  if (tile < 256)      { job = 0; rowbase = tile * 16; }
  else if (tile < 512) { job = 1; rowbase = (tile - 256) * 16; }
  else                 { job = 2; rowbase = (tile - 512) * 16; }
  const EncP& P = (job == 0) ? actorP : encP;
  const int agent_i = agent_i_p[0];
  const int t = threadIdx.x;   // t == output feature h for VALU phases

  const int gcnt[4] = {3, 4, BOXN, RAMPN};
  const int gk  [4] = {10, 10, 12, 12};
  const int goff[4] = {10, 10 + 10*(ADV-1), 10 + 10*(AGENT-2+1) - 10 + 70, 0};
  // explicit offsets per reference slicing:
  //   seeker: 10, hider: 40, box: 80, ramp: 80 + 13*BOXN = 197
  const int goffs[4] = {10, 40, 80, 80 + 13*BOXN};
  (void)goff;

  // ---- emb_self = tanh(x[:, :10] @ self_W + self_b) ----
  {
    float w[10];
#pragma unroll
    for (int k = 0; k < 10; ++k) w[k] = P.selfW[k * Hh + t];
    const float bse = P.selfB[t];
    for (int m = 0; m < 16; ++m) {
      const float* x = row_ptr(job, rowbase, m, inputs, share, agent_i);
      float acc = bse;
#pragma unroll
      for (int k = 0; k < 10; ++k) acc += x[k] * w[k];
      sh_self[m * LDA_H + t] = (_Float16)tanhf(acc);
    }
  }
  __syncthreads();

  // ---- gi = tanh(emb_self @ fc_W + fc_b) -> cat[:, 0:256] ----
  gemm_layer(sh_self, LDA_H, P.fcWt, Hh, Hh, P.fcB, true, 1, sh_cat, LDA_CAT);
  __syncthreads();

  // ---- four attention groups ----
  for (int g = 0; g < 4; ++g) {
    const int ng = gcnt[g], K = gk[g], ioff = goffs[g];
    const float* W = P.gW[g]; const float* Bv = P.gB[g];

    // qc = emb_self @ corr_g (f32)
    gemm_layer(sh_self, LDA_H, P.corr[g], Hh, Hh, nullptr, false, 0, sh_qc, LDA_QC);
    __syncthreads();

    float wreg[12];
    for (int k = 0; k < K; ++k) wreg[k] = W[k * Hh + t];
    const float bent = Bv[t];

    // pass 1: beta[m][e] = qc[m] . emb[m][e]
    for (int e = 0; e < ng; ++e) {
      for (int m = 0; m < 16; ++m) {
        const float* x = row_ptr(job, rowbase, m, inputs, share, agent_i) + ioff + e * K;
        float acc = bent;
        for (int k = 0; k < K; ++k) acc += x[k] * wreg[k];
        sh_ent[m * LDA_H + t] = (_Float16)tanhf(acc);
      }
      __syncthreads();
      {
        int m = t >> 4, j = t & 15;
        float p = 0.f;
#pragma unroll
        for (int hh = 0; hh < 16; ++hh) {
          int h = j * 16 + hh;
          p += sh_qc[m * LDA_QC + h] * (float)sh_ent[m * LDA_H + h];
        }
        sh_red[m * 16 + j] = p;
      }
      __syncthreads();
      if (t < 16) {
        float s = 0.f;
#pragma unroll
        for (int j = 0; j < 16; ++j) s += sh_red[t * 16 + j];
        sh_beta[t * 9 + e] = s;
      }
      __syncthreads();
    }

    // softmax over entities (per row)
    if (t < 16) {
      float mx = -1e30f;
      for (int e = 0; e < ng; ++e) mx = fmaxf(mx, sh_beta[t * 9 + e]);
      float s = 0.f;
      for (int e = 0; e < ng; ++e) {
        float ex = __expf(sh_beta[t * 9 + e] - mx);
        sh_alpha[t * 9 + e] = ex; s += ex;
      }
      float inv = 1.f / s;
      for (int e = 0; e < ng; ++e) sh_alpha[t * 9 + e] *= inv;
    }
    __syncthreads();

    // pass 2: vi[m] = sum_e alpha[m][e] * emb[m][e]  (recompute embs)
    float viacc[16];
#pragma unroll
    for (int m = 0; m < 16; ++m) viacc[m] = 0.f;
    for (int e = 0; e < ng; ++e) {
      for (int m = 0; m < 16; ++m) {
        const float* x = row_ptr(job, rowbase, m, inputs, share, agent_i) + ioff + e * K;
        float acc = bent;
        for (int k = 0; k < K; ++k) acc += x[k] * wreg[k];
        sh_ent[m * LDA_H + t] = (_Float16)tanhf(acc);
      }
      __syncthreads();
#pragma unroll
      for (int m = 0; m < 16; ++m)
        viacc[m] += sh_alpha[m * 9 + e] * (float)sh_ent[m * LDA_H + t];
      __syncthreads();
    }
    const int cbase = (1 + g) * Hh;
#pragma unroll
    for (int m = 0; m < 16; ++m)
      sh_cat[m * LDA_CAT + cbase + t] = (_Float16)viacc[m];
    __syncthreads();
  }

  // ---- h = tanh(cat @ enc1_W + b) ; out = tanh(h @ enc2_W + b) ----
  gemm_layer(sh_cat, LDA_CAT, P.enc1Wt, 5 * Hh, 5 * Hh, P.enc1B, true, 1, sh_h, LDA_H);
  __syncthreads();
  if (job == 0)
    gemm_layer(sh_h, LDA_H, P.enc2Wt, Hh, Hh, P.enc2B, true, 0,
               outHidden + (size_t)rowbase * Hh, Hh);
  else if (job == 1)
    gemm_layer(sh_h, LDA_H, P.enc2Wt, Hh, Hh, P.enc2B, true, 1,
               fii + (size_t)rowbase * Hh, Hh);
  else
    gemm_layer(sh_h, LDA_H, P.enc2Wt, Hh, Hh, P.enc2B, true, 1,
               fij + (size_t)rowbase * Hh, Hh);
}

// ---------------- critic kernel -------------------------------------
__global__ void __launch_bounds__(256)
critic_kernel(const _Float16* __restrict__ fii, const _Float16* __restrict__ fij,
              const _Float16* __restrict__ corrT, const _Float16* __restrict__ cfcT,
              const _Float16* __restrict__ c1T,
              const float* __restrict__ fcB, const float* __restrict__ c1B,
              const float* __restrict__ c2W, const float* __restrict__ c2B,
              const float* __restrict__ rnn, float* __restrict__ out)
{
  __shared__ _Float16 sh_fii[16 * LDA_H];
  __shared__ float    sh_qc [16 * LDA_QC];
  __shared__ _Float16 sh_cat[16 * 520];
  __shared__ _Float16 sh_h  [16 * LDA_H];
  __shared__ float    sh_red[256];
  __shared__ float    sh_beta [16 * NOTH];
  __shared__ float    sh_alpha[16 * NOTH];

  const int rowbase = blockIdx.x * 16;
  const int t = threadIdx.x;

  for (int idx = t; idx < 16 * Hh; idx += 256) {
    int m = idx >> 8, h = idx & 255;
    sh_fii[m * LDA_H + h] = fii[(size_t)(rowbase + m) * Hh + h];
  }
  __syncthreads();

  gemm_layer(sh_fii, LDA_H, corrT, Hh, Hh, nullptr, false, 0, sh_qc, LDA_QC);
  __syncthreads();
  gemm_layer(sh_fii, LDA_H, cfcT, Hh, Hh, fcB, true, 1, sh_cat, 520);
  __syncthreads();

  for (int n = 0; n < NOTH; ++n) {
    {
      int m = t >> 4, j = t & 15;
      const _Float16* fr = fij + ((size_t)(rowbase + m) * NOTH + n) * Hh;
      float p = 0.f;
#pragma unroll
      for (int hh = 0; hh < 16; ++hh) {
        int h = j * 16 + hh;
        p += sh_qc[m * LDA_QC + h] * (float)fr[h];
      }
      sh_red[m * 16 + j] = p;
    }
    __syncthreads();
    if (t < 16) {
      float s = 0.f;
#pragma unroll
      for (int j = 0; j < 16; ++j) s += sh_red[t * 16 + j];
      sh_beta[t * NOTH + n] = s;
    }
    __syncthreads();
  }
  if (t < 16) {
    float mx = -1e30f;
    for (int n = 0; n < NOTH; ++n) mx = fmaxf(mx, sh_beta[t * NOTH + n]);
    float s = 0.f;
    for (int n = 0; n < NOTH; ++n) {
      float ex = __expf(sh_beta[t * NOTH + n] - mx);
      sh_alpha[t * NOTH + n] = ex; s += ex;
    }
    float inv = 1.f / s;
    for (int n = 0; n < NOTH; ++n) sh_alpha[t * NOTH + n] *= inv;
  }
  __syncthreads();

  float viacc[16];
#pragma unroll
  for (int m = 0; m < 16; ++m) viacc[m] = 0.f;
  for (int n = 0; n < NOTH; ++n)
#pragma unroll
    for (int m = 0; m < 16; ++m)
      viacc[m] += sh_alpha[m * NOTH + n] *
                  (float)fij[((size_t)(rowbase + m) * NOTH + n) * Hh + t];
#pragma unroll
  for (int m = 0; m < 16; ++m)
    sh_cat[m * 520 + Hh + t] = (_Float16)viacc[m];
  __syncthreads();

  gemm_layer(sh_cat, 520, c1T, 2 * Hh, 2 * Hh, c1B, true, 1, sh_h, LDA_H);
  __syncthreads();

  if (t < 16) {
    float acc = c2B[0];
    for (int h = 0; h < Hh; ++h) acc += (float)sh_h[t * LDA_H + h] * c2W[h];
    out[rowbase + t] = acc;                       // value
    out[OUT_RNN + rowbase + t] = rnn[rowbase + t]; // rnn passthrough
  }
}

// ---------------- f32 -> f16 transposed weight conversion -----------
// dst[n*K + k] = src[k*N + n], N = 256
__global__ void __launch_bounds__(256)
cvt_kernel(const float* __restrict__ src, _Float16* __restrict__ dst, int K)
{
  int i = blockIdx.x * blockDim.x + threadIdx.x;
  if (i < K * Hh) {
    int n = i / K, k = i - n * K;
    dst[i] = (_Float16)src[(size_t)k * Hh + n];
  }
}

// ---------------- launch --------------------------------------------
extern "C" void kernel_launch(void* const* d_in, const int* in_sizes, int n_in,
                              void* d_out, int out_size, void* d_ws, size_t ws_size,
                              hipStream_t stream)
{
  (void)in_sizes; (void)n_in; (void)out_size; (void)ws_size;
  auto F = [&](int i) { return (const float*)d_in[i]; };
  _Float16* wsH = (_Float16*)d_ws;

  // d_in layout: 0 share, 1 inputs, 2 rnn, 3 masks, 4..23 actor leaves,
  // 24..43 encoder leaves, 44 corr, 45 fcW, 46 fcB, 47 c1W, 48 c1B,
  // 49 c2W, 50 c2B, 51 agent_i, ...
  struct Cvt { int src; size_t dst; int K; };
  const Cvt tab[17] = {
    {4+10, ACT_BASE+SET_ADV, 256}, {4+11, ACT_BASE+SET_GOOD, 256},
    {4+12, ACT_BASE+SET_BOX, 256}, {4+13, ACT_BASE+SET_RAMP, 256},
    {4+14, ACT_BASE+SET_FCW, 256}, {4+16, ACT_BASE+SET_E1W, 1280},
    {4+18, ACT_BASE+SET_E2W, 256},
    {24+10, ENC_BASE+SET_ADV, 256}, {24+11, ENC_BASE+SET_GOOD, 256},
    {24+12, ENC_BASE+SET_BOX, 256}, {24+13, ENC_BASE+SET_RAMP, 256},
    {24+14, ENC_BASE+SET_FCW, 256}, {24+16, ENC_BASE+SET_E1W, 1280},
    {24+18, ENC_BASE+SET_E2W, 256},
    {44, CRT_CORR, 256}, {45, CRT_CFC, 256}, {47, CRT_C1, 512},
  };
  for (int i = 0; i < 17; ++i) {
    int total = tab[i].K * Hh;
    cvt_kernel<<<(total + 255) / 256, 256, 0, stream>>>(F(tab[i].src),
                                                        wsH + tab[i].dst, tab[i].K);
  }

  auto mkP = [&](int b0, size_t wb) {
    EncP P{};
    P.selfW = F(b0 + 0); P.selfB = F(b0 + 1);
    P.gW[0] = F(b0 + 2); P.gB[0] = F(b0 + 3);
    P.gW[1] = F(b0 + 4); P.gB[1] = F(b0 + 5);
    P.gW[2] = F(b0 + 6); P.gB[2] = F(b0 + 7);
    P.gW[3] = F(b0 + 8); P.gB[3] = F(b0 + 9);
    P.fcB   = F(b0 + 15); P.enc1B = F(b0 + 17); P.enc2B = F(b0 + 19);
    P.corr[0] = wsH + wb + SET_ADV;  P.corr[1] = wsH + wb + SET_GOOD;
    P.corr[2] = wsH + wb + SET_BOX;  P.corr[3] = wsH + wb + SET_RAMP;
    P.fcWt   = wsH + wb + SET_FCW;
    P.enc1Wt = wsH + wb + SET_E1W;
    P.enc2Wt = wsH + wb + SET_E2W;
    return P;
  };
  EncP actorP = mkP(4, ACT_BASE);
  EncP encP   = mkP(24, ENC_BASE);

  float* out = (float*)d_out;
  _Float16* fii = wsH + WS_FII_H;
  _Float16* fij = wsH + WS_FIJ_H;

  const int enc_blocks = 256 + 256 + (Bb * NOTH) / 16;  // 2304
  enc_kernel<<<enc_blocks, 256, 0, stream>>>(F(1), F(0), actorP, encP,
                                             (const int*)d_in[51],
                                             out + OUT_HID, fii, fij);

  critic_kernel<<<Bb / 16, 256, 0, stream>>>(fii, fij,
                                             wsH + CRT_CORR, wsH + CRT_CFC,
                                             wsH + CRT_C1,
                                             F(46), F(48), F(49), F(50),
                                             F(2), out);
}
